// AttnDecoderRNN_80264348827963
// MI455X (gfx1250) — matrix operand order repaired and checked by reading, not compiled, not run on hardware
//
#include <hip/hip_runtime.h>
#include <hip/hip_bf16.h>

// ---------------------------------------------------------------------------
// AttnDecoderRNN for MI455X (gfx1250, wave32, WMMA).
// H=1024, V=16000, B=64, L=256, T=128.
//
// - All GEMMs via v_wmma_f32_16x16x32_bf16 (bf16 in, f32 accumulate).
// - Weights (and enc, workspace permitting) converted to bf16 once per
//   launch; steady-state working set (~84MB) fits the 192MB L2.
// - 4 kernels per step (launch latency dominates a sequential recurrence):
//     k_attn       : embed gather + scores + softmax + context   (fused)
//     k_comb       : x = relu([e|ctx] @ W_comb^T + b)            (WMMA)
//     k_gates_cell : gates GEMMs + LSTM cell, LDS-staged          (WMMA)
//     k_logits     : logits GEMM + batch-axis log_softmax         (WMMA)
// - h is double-buffered across steps so the fused gates+cell kernel has no
//   cross-block RAW hazard; c is block-exclusive and updates in place.
// ---------------------------------------------------------------------------

#define H      1024
#define Vv     16000
#define Bsz    64
#define Lmax   256
#define Tsteps 128
#define BH     (Bsz * H)   // 65536

typedef __attribute__((ext_vector_type(16))) __bf16       v16bf;
typedef __attribute__((ext_vector_type(8)))  float        v8f;
typedef __attribute__((ext_vector_type(4)))  unsigned int v4u;

struct Pack32 { v4u lo, hi; };

// float -> bf16 (round-to-nearest-even) on raw ushort storage.
static __device__ __forceinline__ unsigned short f2bf(float f) {
  unsigned int u = __float_as_uint(f);
  unsigned int r = (u + 0x7FFFu + ((u >> 16) & 1u)) >> 16;
  return (unsigned short)r;
}
static __device__ __forceinline__ float bf2f(unsigned short u) {
  return __uint_as_float((unsigned int)u << 16);
}

// 32-byte WMMA operand fragment from two 16B chunks of a row-major bf16
// matrix.  e_lo / e_hi are element (ushort) offsets from p.
static __device__ __forceinline__ v16bf ld_frag(const unsigned short* p,
                                                int e_lo, int e_hi) {
  v4u lo = *(const v4u*)(p + e_lo);
  v4u hi = *(const v4u*)(p + e_hi);
  Pack32 pk{lo, hi};
  return __builtin_bit_cast(v16bf, pk);
}

static __device__ __forceinline__ v8f wmma_bf16(v16bf a, v16bf b, v8f c) {
  return __builtin_amdgcn_wmma_f32_16x16x32_bf16(false, a, false, b,
                                                 (short)0, c, false, false);
}

// ---------------------------------------------------------------------------
// One-time kernels
// ---------------------------------------------------------------------------
__global__ void k_cvt(const float* __restrict__ src,
                      unsigned short* __restrict__ dst, long n) {
  long i = (long)blockIdx.x * blockDim.x + threadIdx.x;
  if (i < n) dst[i] = f2bf(src[i]);
}

__global__ void k_init(const float* __restrict__ enc, float* __restrict__ hf,
                       unsigned short* __restrict__ hb, float* __restrict__ cf) {
  int i = blockIdx.x * 256 + threadIdx.x;        // i < BH
  float v = enc[(size_t)(Lmax - 1) * BH + i];    // dec_h = encoder_outputs[-1]
  hf[i] = v;
  hb[i] = f2bf(v);
  cf[i] = 0.f;
}

// ---------------------------------------------------------------------------
// Step kernel 1 (fused): embed gather + attention scores + softmax + context.
// Block b:  scores[b,l] = b_attn[l] + [e|h][b,:] . W_attn[l,:]
//           aw = softmax_l(scores)  (kept in LDS)
//           ctx[b,k] = sum_l aw[l] * enc[l,b,k]
// Writes [e|ctx] (bf16) for the combine GEMM.  grid = 64 blocks x 256 thr.
// ---------------------------------------------------------------------------
__global__ void k_attn(const int* __restrict__ tgt, int t,
                       const float* __restrict__ emb,
                       const float* __restrict__ hf,
                       const float* __restrict__ Wattn,
                       const float* __restrict__ battn,
                       const float* __restrict__ encf,
                       const unsigned short* __restrict__ encb, int enc_is_bf16,
                       unsigned short* __restrict__ ec) {
  __shared__ float se[H];
  __shared__ float sh[H];
  __shared__ float red[256];
  __shared__ float saw[Lmax];
  const int b = blockIdx.x, tid = threadIdx.x;
  const int tok = tgt[b * Tsteps + t];
  for (int k = tid; k < H; k += 256) {
    float ev = emb[(size_t)tok * H + k];
    se[k] = ev;
    ec[b * (2 * H) + k] = f2bf(ev);              // e half of concat(e, ctx)
    sh[k] = hf[b * H + k];
  }
  __syncthreads();

  // ---- scores (one thread per l) ----
  const float4* w0 = (const float4*)(Wattn + (size_t)tid * (2 * H));
  const float4* w1 = w0 + (H / 4);
  const float4* e4 = (const float4*)se;
  const float4* h4 = (const float4*)sh;
  float acc = battn[tid];
  for (int k = 0; k < H / 4; ++k) {
    float4 wa = w0[k], ea = e4[k];
    acc += wa.x * ea.x + wa.y * ea.y + wa.z * ea.z + wa.w * ea.w;
    float4 wb = w1[k], ha = h4[k];
    acc += wb.x * ha.x + wb.y * ha.y + wb.z * ha.z + wb.w * ha.w;
  }
  // ---- softmax over L=256 ----
  red[tid] = acc; __syncthreads();
  for (int s = 128; s > 0; s >>= 1) {
    if (tid < s) red[tid] = fmaxf(red[tid], red[tid + s]);
    __syncthreads();
  }
  float m = red[0]; __syncthreads();
  float ev = __expf(acc - m);
  red[tid] = ev; __syncthreads();
  for (int s = 128; s > 0; s >>= 1) {
    if (tid < s) red[tid] += red[tid + s];
    __syncthreads();
  }
  saw[tid] = ev / red[0];
  __syncthreads();

  // ---- context: each thread owns k = tid + 256*j, j=0..3 ----
  float cacc[4] = {0.f, 0.f, 0.f, 0.f};
  if (enc_is_bf16) {
    const unsigned short* ep = encb + (size_t)b * H + tid;
    for (int l = 0; l < Lmax; ++l) {
      float s = saw[l];
      const unsigned short* row = ep + (size_t)l * BH;
      cacc[0] += s * bf2f(row[0]);
      cacc[1] += s * bf2f(row[256]);
      cacc[2] += s * bf2f(row[512]);
      cacc[3] += s * bf2f(row[768]);
    }
  } else {
    const float* ep = encf + (size_t)b * H + tid;
    for (int l = 0; l < Lmax; ++l) {
      float s = saw[l];
      const float* row = ep + (size_t)l * BH;
      cacc[0] += s * row[0];
      cacc[1] += s * row[256];
      cacc[2] += s * row[512];
      cacc[3] += s * row[768];
    }
  }
  #pragma unroll
  for (int j = 0; j < 4; ++j)
    ec[b * (2 * H) + H + tid + 256 * j] = f2bf(cacc[j]);
}

// ---------------------------------------------------------------------------
// Step kernel 2 (WMMA): x = relu([e|ctx] @ W_comb^T + b_comb) -> bf16
// out [64,1024], K = 2048.  grid.x = 16 (64-col tiles), 8 waves as 4x2.
// ---------------------------------------------------------------------------
__global__ void k_comb(const unsigned short* __restrict__ ec,
                       const unsigned short* __restrict__ Wc,
                       const float* __restrict__ bcomb,
                       unsigned short* __restrict__ xout) {
  const int lane = threadIdx.x & 31, wave = threadIdx.x >> 5;
  const int half = lane >> 4, ln = lane & 15;
  const int row_base = (wave >> 1) * 16;
  const int cb = blockIdx.x * 64 + (wave & 1) * 32;
  const unsigned short* ar = ec + (size_t)(row_base + ln) * (2 * H);
  const unsigned short* b0 = Wc + (size_t)(cb + ln) * (2 * H);
  const unsigned short* b1 = Wc + (size_t)(cb + 16 + ln) * (2 * H);
  v8f acc0 = {}, acc1 = {};
  for (int k0 = 0; k0 < 2 * H; k0 += 32) {
    v16bf a  = ld_frag(ar, k0 + 16 * half, k0 + 16 * half + 8);
    v16bf f0 = ld_frag(b0, k0 + 16 * half, k0 + 16 * half + 8);
    v16bf f1 = ld_frag(b1, k0 + 16 * half, k0 + 16 * half + 8);
    acc0 = wmma_bf16(a, f0, acc0);
    acc1 = wmma_bf16(a, f1, acc1);
  }
  const int c0 = cb + ln, c1 = cb + 16 + ln;
  const float bias0 = bcomb[c0], bias1 = bcomb[c1];
  #pragma unroll
  for (int v = 0; v < 8; ++v) {
    int row = row_base + v + 8 * half;
    xout[row * H + c0] = f2bf(fmaxf(acc0[v] + bias0, 0.f));
    xout[row * H + c1] = f2bf(fmaxf(acc1[v] + bias1, 0.f));
  }
}

// ---------------------------------------------------------------------------
// Step kernel 3 (WMMA, fused): gates = x@W_ih^T + h@W_hh^T + b_ih + b_hh,
// then the LSTM cell.  Block j owns batch x cols {n0..n0+63} of ALL FOUR
// gates (i at n0, f at n0+H, g at n0+2H, o at n0+3H) so the cell is
// block-local.  Gate tile staged in LDS (64x256 f32 = 64KB).
// Logical col lc in [0,256): gate group = lc>>6, global col =
// (lc>>6)*H + n0 + (lc&63).  Wave w: rows (w>>1)*16, lc tiles (w&1)*8..+8.
// grid.x = 16, 8 waves.
// ---------------------------------------------------------------------------
__global__ void k_gates_cell(const unsigned short* __restrict__ xb,
                             const unsigned short* __restrict__ hbo,
                             const unsigned short* __restrict__ Wih,
                             const unsigned short* __restrict__ Whh,
                             const float* __restrict__ bih,
                             const float* __restrict__ bhh,
                             float* __restrict__ cf,
                             float* __restrict__ hfn,
                             unsigned short* __restrict__ hbn) {
  __shared__ float sgt[64][256];                 // 64KB gate tile
  const int lane = threadIdx.x & 31, wave = threadIdx.x >> 5;
  const int half = lane >> 4, ln = lane & 15;
  const int row_base = (wave >> 1) * 16;
  const int n0 = blockIdx.x * 64;
  const int lt0 = (wave & 1) * 8;                // first of 8 lc-tiles

  v8f acc[8];
  #pragma unroll
  for (int j = 0; j < 8; ++j) acc[j] = v8f{};

  const unsigned short* bp[8];
  int gcol[8];
  #pragma unroll
  for (int j = 0; j < 8; ++j) {
    int lt = lt0 + j;
    gcol[j] = (lt >> 2) * H + n0 + (lt & 3) * 16 + ln;   // global gate col
  }

  // pass 1: x @ W_ih^T
  {
    const unsigned short* ar = xb + (size_t)(row_base + ln) * H;
    #pragma unroll
    for (int j = 0; j < 8; ++j) bp[j] = Wih + (size_t)gcol[j] * H;
    for (int k0 = 0; k0 < H; k0 += 32) {
      v16bf a = ld_frag(ar, k0 + 16 * half, k0 + 16 * half + 8);
      #pragma unroll
      for (int j = 0; j < 8; ++j) {
        v16bf f = ld_frag(bp[j], k0 + 16 * half, k0 + 16 * half + 8);
        acc[j] = wmma_bf16(a, f, acc[j]);
      }
    }
  }
  // pass 2: h @ W_hh^T
  {
    const unsigned short* ar = hbo + (size_t)(row_base + ln) * H;
    #pragma unroll
    for (int j = 0; j < 8; ++j) bp[j] = Whh + (size_t)gcol[j] * H;
    for (int k0 = 0; k0 < H; k0 += 32) {
      v16bf a = ld_frag(ar, k0 + 16 * half, k0 + 16 * half + 8);
      #pragma unroll
      for (int j = 0; j < 8; ++j) {
        v16bf f = ld_frag(bp[j], k0 + 16 * half, k0 + 16 * half + 8);
        acc[j] = wmma_bf16(a, f, acc[j]);
      }
    }
  }
  // bias + stage to LDS
  #pragma unroll
  for (int j = 0; j < 8; ++j) {
    int lt = lt0 + j;
    int lc = lt * 16 + ln;
    float bias = bih[gcol[j]] + bhh[gcol[j]];
    #pragma unroll
    for (int v = 0; v < 8; ++v) {
      int row = row_base + v + 8 * half;
      sgt[row][lc] = acc[j][v] + bias;
    }
  }
  __syncthreads();

  // LSTM cell: 64 rows x 64 local k -> 16 cells per thread
  for (int idx = threadIdx.x; idx < 64 * 64; idx += 256) {
    int row = idx >> 6, kl = idx & 63;
    float ig = sgt[row][kl];
    float fg = sgt[row][64 + kl];
    float gg = sgt[row][128 + kl];
    float og = sgt[row][192 + kl];
    float si = 1.f / (1.f + __expf(-ig));
    float sf = 1.f / (1.f + __expf(-fg));
    float so = 1.f / (1.f + __expf(-og));
    int i = row * H + n0 + kl;                   // block-exclusive c slice
    float cn = sf * cf[i] + si * tanhf(gg);
    float hn = so * tanhf(cn);
    cf[i] = cn;
    hfn[i] = hn;
    hbn[i] = f2bf(hn);
  }
}

// ---------------------------------------------------------------------------
// Step kernel 4 (WMMA): logits = h @ W_out^T + b_out, then log_softmax over
// the BATCH axis (axis 0) — each block owns all 64 rows of a 64-col vocab
// tile, so the reduction is block-local in LDS.  grid.x = 250.
// ---------------------------------------------------------------------------
__global__ void k_logits(const unsigned short* __restrict__ hb,
                         const unsigned short* __restrict__ Wo,
                         const float* __restrict__ bout,
                         float* __restrict__ out, int t) {
  __shared__ float tile[64][65];
  __shared__ float lse[64];
  const int lane = threadIdx.x & 31, wave = threadIdx.x >> 5;
  const int half = lane >> 4, ln = lane & 15;
  const int row_base = (wave >> 1) * 16;
  const int colbase0 = blockIdx.x * 64;
  const int cb = colbase0 + (wave & 1) * 32;
  const unsigned short* ar = hb + (size_t)(row_base + ln) * H;
  const unsigned short* b0 = Wo + (size_t)(cb + ln) * H;
  const unsigned short* b1 = Wo + (size_t)(cb + 16 + ln) * H;
  v8f acc0 = {}, acc1 = {};
  for (int k0 = 0; k0 < H; k0 += 32) {
    v16bf a  = ld_frag(ar, k0 + 16 * half, k0 + 16 * half + 8);
    v16bf f0 = ld_frag(b0, k0 + 16 * half, k0 + 16 * half + 8);
    v16bf f1 = ld_frag(b1, k0 + 16 * half, k0 + 16 * half + 8);
    acc0 = wmma_bf16(a, f0, acc0);
    acc1 = wmma_bf16(a, f1, acc1);
  }
  const int c0l = (wave & 1) * 32 + ln, c1l = c0l + 16;
  const float bias0 = bout[colbase0 + c0l], bias1 = bout[colbase0 + c1l];
  #pragma unroll
  for (int v = 0; v < 8; ++v) {
    int row = row_base + v + 8 * half;
    tile[row][c0l] = acc0[v] + bias0;
    tile[row][c1l] = acc1[v] + bias1;
  }
  __syncthreads();
  if (threadIdx.x < 64) {
    int c = threadIdx.x;
    float m = -3.402823466e38f;
    #pragma unroll 4
    for (int r = 0; r < 64; ++r) m = fmaxf(m, tile[r][c]);
    float s = 0.f;
    #pragma unroll 4
    for (int r = 0; r < 64; ++r) s += __expf(tile[r][c] - m);
    lse[c] = m + __logf(s);
  }
  __syncthreads();
  const size_t obase = (size_t)t * Bsz * Vv + colbase0;
  for (int idx = threadIdx.x; idx < 64 * 64; idx += 256) {
    int r = idx >> 6, c = idx & 63;
    out[obase + (size_t)r * Vv + c] = tile[r][c] - lse[c];
  }
}

// ---------------------------------------------------------------------------
// Launch
// ---------------------------------------------------------------------------
extern "C" void kernel_launch(void* const* d_in, const int* in_sizes, int n_in,
                              void* d_out, int out_size, void* d_ws, size_t ws_size,
                              hipStream_t stream) {
  (void)in_sizes; (void)n_in; (void)out_size;
  const int*   tgt   = (const int*)  d_in[0];   // [B,T]
  const float* enc   = (const float*)d_in[1];   // [L,B,H]
  const float* emb   = (const float*)d_in[2];   // [V,H]
  const float* Wattn = (const float*)d_in[3];   // [L,2H]
  const float* battn = (const float*)d_in[4];   // [L]
  const float* Wcomb = (const float*)d_in[5];   // [H,2H]
  const float* bcomb = (const float*)d_in[6];   // [H]
  const float* Wih   = (const float*)d_in[7];   // [4H,H]
  const float* Whh   = (const float*)d_in[8];   // [4H,H]
  const float* bih   = (const float*)d_in[9];   // [4H]
  const float* bhh   = (const float*)d_in[10];  // [4H]
  const float* Wout  = (const float*)d_in[11];  // [V,H]
  const float* bout  = (const float*)d_in[12];  // [V]
  float* out = (float*)d_out;                   // [T,B,V]

  char* ws = (char*)d_ws;
  size_t off = 0;
  auto alloc = [&](size_t bytes) -> void* {
    void* p = ws + off;
    off += (bytes + 255) & ~(size_t)255;
    return p;
  };
  unsigned short* Wc_b  = (unsigned short*)alloc((size_t)H * 2 * H * 2);   //  4 MB
  unsigned short* Wih_b = (unsigned short*)alloc((size_t)4 * H * H * 2);   //  8 MB
  unsigned short* Whh_b = (unsigned short*)alloc((size_t)4 * H * H * 2);   //  8 MB
  unsigned short* Wo_b  = (unsigned short*)alloc((size_t)Vv * H * 2);      // 32 MB
  float*          hf0   = (float*)alloc((size_t)BH * 4);
  float*          hf1   = (float*)alloc((size_t)BH * 4);
  float*          cf    = (float*)alloc((size_t)BH * 4);
  unsigned short* hb0   = (unsigned short*)alloc((size_t)BH * 2);
  unsigned short* hb1   = (unsigned short*)alloc((size_t)BH * 2);
  unsigned short* ecb   = (unsigned short*)alloc((size_t)Bsz * 2 * H * 2); // [e|ctx]
  unsigned short* xb    = (unsigned short*)alloc((size_t)BH * 2);
  unsigned short* encb  = (unsigned short*)alloc((size_t)Lmax * BH * 2);   // 32 MB
  const int enc_bf = (off <= ws_size) ? 1 : 0;   // fall back to f32 enc if tight

  // One-time conversions (reads stay hot in L2 afterwards).
  {
    long n;
    n = (long)H * 2 * H; k_cvt<<<(unsigned)((n + 255) / 256), 256, 0, stream>>>(Wcomb, Wc_b, n);
    n = (long)4 * H * H; k_cvt<<<(unsigned)((n + 255) / 256), 256, 0, stream>>>(Wih,  Wih_b, n);
    n = (long)4 * H * H; k_cvt<<<(unsigned)((n + 255) / 256), 256, 0, stream>>>(Whh,  Whh_b, n);
    n = (long)Vv * H;    k_cvt<<<(unsigned)((n + 255) / 256), 256, 0, stream>>>(Wout, Wo_b,  n);
    if (enc_bf) {
      n = (long)Lmax * BH;
      k_cvt<<<(unsigned)((n + 255) / 256), 256, 0, stream>>>(enc, encb, n);
    }
  }
  k_init<<<BH / 256, 256, 0, stream>>>(enc, hf0, hb0, cf);

  for (int t = 0; t < Tsteps; ++t) {
    float*          hfo = (t & 1) ? hf1 : hf0;   // old h (f32)
    float*          hfn = (t & 1) ? hf0 : hf1;   // new h (f32)
    unsigned short* hbo = (t & 1) ? hb1 : hb0;   // old h (bf16)
    unsigned short* hbn = (t & 1) ? hb0 : hb1;   // new h (bf16)
    k_attn<<<Bsz, 256, 0, stream>>>(tgt, t, emb, hfo, Wattn, battn,
                                    enc, encb, enc_bf, ecb);
    k_comb<<<H / 64, 256, 0, stream>>>(ecb, Wc_b, bcomb, xb);
    k_gates_cell<<<16, 256, 0, stream>>>(xb, hbo, Wih_b, Whh_b, bih, bhh,
                                         cf, hfn, hbn);
    k_logits<<<Vv / 64, 256, 0, stream>>>(hbn, Wo_b, bout, out, t);
  }
}
